// DeformableMultiHeadedAttention_20383914787375
// MI455X (gfx1250) — matrix operand order repaired and checked by reading, compiled, not run on hardware
//
#include <hip/hip_runtime.h>
#include <math.h>

// ---------------------------------------------------------------------------
// DeformableMultiHeadedAttention for MI455X (gfx1250, wave32, WMMA)
// B=4, M=2048, D=1024, H=16, NK=5, HD=64
// ---------------------------------------------------------------------------

typedef _Float16 half_t;
typedef __attribute__((ext_vector_type(16))) _Float16 v16h;
typedef __attribute__((ext_vector_type(8)))  _Float16 v8h;
typedef __attribute__((ext_vector_type(2)))  _Float16 v2h;
typedef __attribute__((ext_vector_type(8)))  float    v8f;

#define B_  4
#define M_  2048
#define D_  1024
#define H_  16
#define NK_ 5
#define HD_ 64

// ---------------------------------------------------------------------------
// fp32 -> fp16 elementwise conversion (grid-stride)
// ---------------------------------------------------------------------------
__global__ void f32_to_f16_kernel(const float* __restrict__ in,
                                  half_t* __restrict__ out, size_t n) {
    size_t i = (size_t)blockIdx.x * blockDim.x + threadIdx.x;
    size_t stride = (size_t)gridDim.x * blockDim.x;
    for (; i < n; i += stride) out[i] = (half_t)in[i];
}

// ---------------------------------------------------------------------------
// sign_lens[b] = sum_m (1 - mask[b,m]) - 1
// ---------------------------------------------------------------------------
__global__ void signlen_kernel(const int* __restrict__ mask,
                               float* __restrict__ sl, int Mn) {
    __shared__ int sbuf[256];
    const int b = blockIdx.x;
    const int t = threadIdx.x;
    int s = 0;
    for (int i = t; i < Mn; i += 256) s += 1 - mask[(size_t)b * Mn + i];
    sbuf[t] = s;
    __syncthreads();
    for (int o = 128; o > 0; o >>= 1) {
        if (t < o) sbuf[t] += sbuf[t + o];
        __syncthreads();
    }
    if (t == 0) sl[b] = (float)sbuf[0] - 1.0f;
}

// ---------------------------------------------------------------------------
// B fragment from LDS: 32x16 f16. lane idx = column; element j = K(hi*8+j),
// element j+8 = K(16+hi*8+j).
// ---------------------------------------------------------------------------
__device__ __forceinline__ v16h load_bfrag(const half_t* curB, int c,
                                           int idx, int hi) {
    const half_t* pb = curB + (c * 16 + idx) * 32 + hi * 8;
    v8h b0 = *(const v8h*)pb;
    v8h b1 = *(const v8h*)(pb + 16);
    v16h f;
#pragma unroll
    for (int j = 0; j < 8; ++j) { f[j] = b0[j]; f[8 + j] = b1[j]; }
    return f;
}

// ---------------------------------------------------------------------------
// WMMA GEMM: C[rows, cols] = A[rows, kd] (f16 row-major)
//                           x W[cols, kd]^T (f16 row-major) + bias
// Block: 256 threads (8 waves). Block tile 128x128; wave tile 16x128.
// W tile staged global->LDS with GLOBAL_LOAD_ASYNC_TO_LDS_B128 (ASYNCcnt),
// double-buffered so the DMA of tile kb+1 overlaps the 8 WMMAs of tile kb.
// B fragments fetched one stage ahead in a rotating register pair so each
// WMMA only waits on loads issued one fragment earlier.
// One barrier per K-step.  f32 accumulate, f16 or f32 store.
// ---------------------------------------------------------------------------
template <bool OUT_HALF>
__global__ __launch_bounds__(256)
void gemm_wmma_kernel(const half_t* __restrict__ A,
                      const half_t* __restrict__ W,
                      const float* __restrict__ bias,
                      float* __restrict__ outF,
                      half_t* __restrict__ outH,
                      int rows, int cols, int kd) {
    __shared__ __align__(16) half_t ldsB[2][128 * 32];   // 2 x 8 KB

    const int tid  = threadIdx.x;
    const int lane = tid & 31;
    const int wave = tid >> 5;
    const int idx  = lane & 15;   // row (A) / col (B) within fragment
    const int hi   = lane >> 4;   // lane-half selector

    const int colBase = blockIdx.x * 128;
    const int rowBase = blockIdx.y * 128 + wave * 16;

    v8f acc[8];
#pragma unroll
    for (int c = 0; c < 8; ++c)
#pragma unroll
        for (int r = 0; r < 8; ++r) acc[c][r] = 0.0f;

    const int brow = tid >> 1;         // 0..127 : which W row this thread stages
    const int bseg = (tid & 1) * 16;   // 0 / 16 : which 16-half segment
    const int nkt  = kd >> 5;          // K-steps of 32

    // Clamp out-of-range W rows instead of zero-filling: those B columns map
    // 1:1 to C columns >= cols, which the epilogue drops anyway.
    int gcol = colBase + brow;
    if (gcol >= cols) gcol = cols - 1;
    const half_t* wrow = W + (size_t)gcol * kd + bseg;

    // Wave-relative LDS byte offsets (shared aperture lives in addr[63:32]).
    const unsigned dst0 = (unsigned)(size_t)&ldsB[0][brow * 32 + bseg];
    const unsigned dst1 = (unsigned)(size_t)&ldsB[1][brow * 32 + bseg];

    // Preload K-step 0 into buffer 0 (async DMA; INST_OFFSET applies to both
    // the LDS and the global address, so offset:16 covers the second half).
    asm volatile("global_load_async_to_lds_b128 %0, %1, off"
                 :: "v"(dst0), "v"(wrow) : "memory");
    asm volatile("global_load_async_to_lds_b128 %0, %1, off offset:16"
                 :: "v"(dst0), "v"(wrow) : "memory");

    for (int kb = 0; kb < nkt; ++kb) {
        const int k0 = kb << 5;

        // Own async loads done, then block-wide barrier => tile fully in LDS.
        asm volatile("s_wait_asynccnt 0x0" ::: "memory");
        __syncthreads();

        // Kick off the DMA for the next K-step into the other buffer.
        if (kb + 1 < nkt) {
            const half_t* nsrc = wrow + (size_t)(kb + 1) * 32;
            const unsigned ndst = (kb & 1) ? dst0 : dst1;
            asm volatile("global_load_async_to_lds_b128 %0, %1, off"
                         :: "v"(ndst), "v"(nsrc) : "memory");
            asm volatile("global_load_async_to_lds_b128 %0, %1, off offset:16"
                         :: "v"(ndst), "v"(nsrc) : "memory");
        }

        const half_t* curB = &ldsB[kb & 1][0];

        // A fragment: 16x32 f16.  lane idx = row; element j   = K(hi*8+j),
        //                                         element j+8 = K(16+hi*8+j)
        const half_t* pa = A + (size_t)(rowBase + idx) * kd + k0 + hi * 8;
        v8h a0 = *(const v8h*)pa;
        v8h a1 = *(const v8h*)(pa + 16);
        v16h af;
#pragma unroll
        for (int j = 0; j < 8; ++j) { af[j] = a0[j]; af[8 + j] = a1[j]; }

        if (kb + 1 < nkt) __builtin_prefetch(pa + 32, 0, 1);

        // Two-stage rotating pipeline: fetch B fragment c+1 while the XDL
        // pipe chews on fragment c, so the wait before each WMMA is on loads
        // issued one stage earlier (dscnt<=2), not the immediately preceding
        // pair (dscnt==0).
        v16h bcur = load_bfrag(curB, 0, idx, hi);
#pragma unroll
        for (int c = 0; c < 8; ++c) {
            v16h bnxt = bcur;
            if (c < 7) bnxt = load_bfrag(curB, c + 1, idx, hi);
            acc[c] = __builtin_amdgcn_wmma_f32_16x16x32_f16(
                false, af, false, bcur, (short)0, acc[c], false, false);
            bcur = bnxt;
        }
    }

    // Epilogue: C/D layout — lane L, element r -> row (L>>4)*8 + r, col L&15
#pragma unroll
    for (int c = 0; c < 8; ++c) {
        int col = colBase + c * 16 + idx;
        if (col >= cols) continue;
        float bv = bias ? bias[col] : 0.0f;
#pragma unroll
        for (int r = 0; r < 8; ++r) {
            int row = rowBase + hi * 8 + r;
            float val = acc[c][r] + bv;
            size_t o = (size_t)row * cols + col;
            if (OUT_HALF) outH[o] = (half_t)val;
            else          outF[o] = val;
        }
    }
}

// ---------------------------------------------------------------------------
// Deformable gather + attention.  One wave per (b, h, q).
// Each lane owns 2 of the 64 head-dims.  NK=5 softmax in registers.
// K/V f16 are L2-resident (32 MB << 192 MB), so gathers are L2-served.
// ---------------------------------------------------------------------------
__global__ __launch_bounds__(256)
void deform_attn_kernel(const half_t* __restrict__ Kp,
                        const half_t* __restrict__ Vp,
                        const half_t* __restrict__ Qp,
                        const float* __restrict__ Off,
                        const float* __restrict__ sl,
                        half_t* __restrict__ ctx) {
    const int lane = threadIdx.x & 31;
    const int wave = threadIdx.x >> 5;
    const int gid  = blockIdx.x * 8 + wave;        // (b*H + h)*M + q
    const int q  = gid % M_;
    const int bh = gid / M_;
    const int h  = bh % H_;
    const int b  = bh / H_;

    const size_t rowQ    = (size_t)b * M_ + q;
    const size_t headOff = (size_t)h * HD_;

    v2h qv = *(const v2h*)(Qp + rowQ * D_ + headOff + lane * 2);
    const float qa = (float)qv[0] * 0.125f;        // 1/sqrt(64)
    const float qb = (float)qv[1] * 0.125f;

    const float slen = sl[b];
    const float loc  = (float)q;
    const float* offp = Off + rowQ * (H_ * NK_) + h * NK_;

    float sva[NK_], svb[NK_], sc[NK_];

#pragma unroll
    for (int nk = 0; nk < NK_; ++nk) {
        float x = (float)(nk - 3) + offp[nk] + loc;   // ref = [-3..1]
        float lx = fmodf(x, slen);                    // python-style mod
        if (lx != 0.0f && ((lx < 0.0f) != (slen < 0.0f))) lx += slen;
        float xg = lx / (float)(M_ - 1) * 2.0f - 1.0f;
        float ix = ((xg + 1.0f) * (float)M_ - 1.0f) * 0.5f;
        float x0 = floorf(ix);
        float dx = ix - x0;
        int i0 = (int)x0;
        int i1 = i0 + 1;
        float w0 = 1.0f - dx, w1 = dx;

        float k0a = 0.f, k0b = 0.f, k1a = 0.f, k1b = 0.f;
        float v0a = 0.f, v0b = 0.f, v1a = 0.f, v1b = 0.f;
        if (i0 >= 0 && i0 < M_) {
            size_t r = ((size_t)b * M_ + i0) * D_ + headOff + lane * 2;
            v2h kk = *(const v2h*)(Kp + r); k0a = (float)kk[0]; k0b = (float)kk[1];
            v2h vv = *(const v2h*)(Vp + r); v0a = (float)vv[0]; v0b = (float)vv[1];
        }
        if (i1 >= 0 && i1 < M_) {
            size_t r = ((size_t)b * M_ + i1) * D_ + headOff + lane * 2;
            v2h kk = *(const v2h*)(Kp + r); k1a = (float)kk[0]; k1b = (float)kk[1];
            v2h vv = *(const v2h*)(Vp + r); v1a = (float)vv[0]; v1b = (float)vv[1];
        }
        float ska = 0.5f * (w0 * k0a + w1 * k1a);
        float skb = 0.5f * (w0 * k0b + w1 * k1b);
        sva[nk]   = 0.5f * (w0 * v0a + w1 * v1a);
        svb[nk]   = 0.5f * (w0 * v0b + w1 * v1b);

        float s = qa * ska + qb * skb;
#pragma unroll
        for (int o = 16; o > 0; o >>= 1) s += __shfl_xor(s, o, 32);
        sc[nk] = s;
    }

    float mx = sc[0];
#pragma unroll
    for (int nk = 1; nk < NK_; ++nk) mx = fmaxf(mx, sc[nk]);
    float e[NK_], den = 0.0f;
#pragma unroll
    for (int nk = 0; nk < NK_; ++nk) { e[nk] = __expf(sc[nk] - mx); den += e[nk]; }
    float inv = 1.0f / den;

    float ca = 0.0f, cb = 0.0f;
#pragma unroll
    for (int nk = 0; nk < NK_; ++nk) {
        float a = e[nk] * inv;
        ca += a * sva[nk];
        cb += a * svb[nk];
    }

    v2h o;
    o[0] = (half_t)ca;
    o[1] = (half_t)cb;
    *(v2h*)(ctx + rowQ * D_ + headOff + lane * 2) = o;
}

// ---------------------------------------------------------------------------
// Launch
// ---------------------------------------------------------------------------
extern "C" void kernel_launch(void* const* d_in, const int* in_sizes, int n_in,
                              void* d_out, int out_size, void* d_ws, size_t ws_size,
                              hipStream_t stream) {
    const float* k    = (const float*)d_in[0];
    const float* v    = (const float*)d_in[1];
    const float* q    = (const float*)d_in[2];
    const int*   mask = (const int*)  d_in[3];
    const float* Wk   = (const float*)d_in[4];
    const float* bk   = (const float*)d_in[5];
    const float* Wv   = (const float*)d_in[6];
    const float* bv   = (const float*)d_in[7];
    const float* Wq   = (const float*)d_in[8];
    const float* bq   = (const float*)d_in[9];
    const float* Woff = (const float*)d_in[10];
    const float* boff = (const float*)d_in[11];
    const float* Wo   = (const float*)d_in[12];
    const float* bo   = (const float*)d_in[13];
    float* out = (float*)d_out;

    const size_t rows  = (size_t)B_ * M_;       // 8192
    const size_t actN  = rows * D_;             // 8,388,608
    const size_t wN    = (size_t)D_ * D_;       // 1,048,576
    const size_t woffN = (size_t)H_ * NK_ * D_; // 81,920

    char* ws = (char*)d_ws;
    size_t off = 0;
    auto alloc = [&](size_t bytes) -> char* {
        char* p = ws + off;
        off += (bytes + 255) & ~(size_t)255;
        return p;
    };
    half_t* k16    = (half_t*)alloc(actN * 2);
    half_t* v16    = (half_t*)alloc(actN * 2);
    half_t* q16    = (half_t*)alloc(actN * 2);
    half_t* Wk16   = (half_t*)alloc(wN * 2);
    half_t* Wv16   = (half_t*)alloc(wN * 2);
    half_t* Wq16   = (half_t*)alloc(wN * 2);
    half_t* Wo16   = (half_t*)alloc(wN * 2);
    half_t* Woff16 = (half_t*)alloc(woffN * 2);
    half_t* Kp16   = (half_t*)alloc(actN * 2);
    half_t* Vp16   = (half_t*)alloc(actN * 2);
    half_t* Qp16   = (half_t*)alloc(actN * 2);
    float*  Offb   = (float*) alloc(rows * H_ * NK_ * 4);
    half_t* ctx16  = (half_t*)alloc(actN * 2);
    float*  sl     = (float*) alloc(B_ * 4);

    // 1) fp32 -> fp16 staging
    f32_to_f16_kernel<<<4096, 256, 0, stream>>>(k,    k16,    actN);
    f32_to_f16_kernel<<<4096, 256, 0, stream>>>(v,    v16,    actN);
    f32_to_f16_kernel<<<4096, 256, 0, stream>>>(q,    q16,    actN);
    f32_to_f16_kernel<<<1024, 256, 0, stream>>>(Wk,   Wk16,   wN);
    f32_to_f16_kernel<<<1024, 256, 0, stream>>>(Wv,   Wv16,   wN);
    f32_to_f16_kernel<<<1024, 256, 0, stream>>>(Wq,   Wq16,   wN);
    f32_to_f16_kernel<<<1024, 256, 0, stream>>>(Wo,   Wo16,   wN);
    f32_to_f16_kernel<<<128,  256, 0, stream>>>(Woff, Woff16, woffN);

    // 2) sign_lens
    signlen_kernel<<<B_, 256, 0, stream>>>(mask, sl, M_);

    // 3) projections (WMMA f16 -> f16 with f32 accumulate)
    dim3 blk(256);
    dim3 gD((unsigned)(D_ / 128), (unsigned)(rows / 128));  // (8, 64)
    gemm_wmma_kernel<true><<<gD, blk, 0, stream>>>(k16, Wk16, bk, nullptr, Kp16,
                                                   (int)rows, D_, D_);
    gemm_wmma_kernel<true><<<gD, blk, 0, stream>>>(v16, Wv16, bv, nullptr, Vp16,
                                                   (int)rows, D_, D_);
    gemm_wmma_kernel<true><<<gD, blk, 0, stream>>>(q16, Wq16, bq, nullptr, Qp16,
                                                   (int)rows, D_, D_);

    // 4) offsets from projected q: [rows, 80] fp32
    dim3 gOff(1u, (unsigned)(rows / 128));
    gemm_wmma_kernel<false><<<gOff, blk, 0, stream>>>(Qp16, Woff16, boff, Offb,
                                                      nullptr, (int)rows,
                                                      H_ * NK_, D_);

    // 5) deformable gather + softmax + context (one wave per (b,h,q))
    deform_attn_kernel<<<(B_ * H_ * M_) / 8, 256, 0, stream>>>(
        Kp16, Vp16, Qp16, Offb, sl, ctx16);

    // 6) output projection -> fp32 d_out
    gemm_wmma_kernel<false><<<gD, blk, 0, stream>>>(ctx16, Wo16, bo, out, nullptr,
                                                    (int)rows, D_, D_);
}
